// SparsTriangularUpdate_43782896615943
// MI455X (gfx1250) — compile-verified, matrix operand order checked by reading, not run on hardware
//
#include <hip/hip_runtime.h>

// ---------------------------------------------------------------------------
// Triangular multiplicative update (AlphaFold-style), MI455X / gfx1250.
// B=1, L=512, D=C=128.
//   K1 ln_proj : zn = LN(z); a,b = gated projections; gate = sigmoid(zn@Wgo+bgo)
//                a stored channel-major [c][i][m]; b stored channel-major
//                TRANSPOSED [c][j][m] so kernel-2's B operand is K-contiguous.
//   K2 tri_gemm: k[i,j,c] = sum_m aT[c][i][m]*bTT[c][j][m] (bf16 WMMA, f32 acc)
//                A/B K-slabs staged via the Tensor Data Mover (tensor_load_to_lds,
//                TENSORcnt-tracked, double-buffered LDS) issued by wave 0.
//   K3 out_proj: out = gate * (LN_c(k) @ Wlo + blo)        (f32 out)
// All operands are laid out so every WMMA fragment is exactly two contiguous
// 16-byte LDS loads per lane (ds_load_b128), matching the wave32 fragment
// layout: lane&15 = row/col index, lane>=16 carries K phases +8/+24.
// ---------------------------------------------------------------------------

typedef __attribute__((ext_vector_type(16))) __bf16 v16bf;
typedef __attribute__((ext_vector_type(8)))  float  v8f;
typedef __attribute__((ext_vector_type(4)))  unsigned int v4u;
typedef __attribute__((ext_vector_type(8)))  int v8i;
typedef __attribute__((ext_vector_type(4)))  int v4i;

#define LSEQ   512
#define DIM    128
#define NTOT   (LSEQ * LSEQ)          // 262144 rows

// workspace layout (bytes)
#define OFF_AT   ((size_t)0)                    // 128 * 512*512 bf16 = 64 MB
#define OFF_BT   ((size_t)64 * 1024 * 1024)     // 64 MB
#define OFF_GATE ((size_t)128 * 1024 * 1024)    // 64 MB
#define OFF_K    ((size_t)192 * 1024 * 1024)    // 64 MB

#if defined(__has_builtin)
#if __has_builtin(__builtin_amdgcn_tensor_load_to_lds)
#define HAVE_TDM 1
#endif
#endif

__device__ __forceinline__ float sigm(float x) {
  return 1.0f / (1.0f + __expf(-x));
}

// Fragment load from K-contiguous (row-major [row][k]) bf16 source.
// Two 16-byte LDS loads per lane, no element shuffling.
__device__ __forceinline__ v16bf frag_rm(const __bf16* base, int ld, int row0,
                                         int k0, int lane) {
  const int m  = lane & 15;
  const int kq = (lane >> 4) * 8;
  const __bf16* p = base + (size_t)(row0 + m) * ld + k0 + kq;
  union { v16bf v; uint4 q[2]; } u;
  u.q[0] = *(const uint4*)(p);
  u.q[1] = *(const uint4*)(p + 16);
  return u.v;
}

#define WMMA_BF16(A, B, C) \
  __builtin_amdgcn_wmma_f32_16x16x32_bf16(false, (A), false, (B), (short)0, (C), false, false)

#ifdef HAVE_TDM
// --- Tensor DMA descriptor construction (CDNA5 ISA ch.8) -------------------
// Group 0: [1:0]=count, [63:32]=lds_addr, [120:64]=global_addr, [127:126]=type(2)
__device__ __forceinline__ v4u tdm_group0(unsigned lds_off,
                                          unsigned long long gaddr) {
  v4u g;
  g[0] = 1u;
  g[1] = lds_off;
  g[2] = (unsigned)(gaddr & 0xffffffffull);
  g[3] = (unsigned)((gaddr >> 32) & 0x1ffffffull) | (2u << 30);
  return g;
}
// Group 1 for a 2D bf16 tile: data_size=1 (2 bytes), tile_dim0 contiguous
// elements per row, tile_dim1 rows, row pitch tensor_dim0_stride elements.
__device__ __forceinline__ v8i tdm_group1_2d(unsigned tile_d0, unsigned tile_d1,
                                             unsigned tensor_d0, unsigned tensor_d1,
                                             unsigned long long stride0) {
  v8i g;
  g[0] = (int)(1u << 16);                                     // data_size = 2B
  g[1] = (int)((tensor_d0 & 0xffffu) << 16);                  // tensor_dim0 lo
  g[2] = (int)((tensor_d0 >> 16) | ((tensor_d1 & 0xffffu) << 16));
  g[3] = (int)((tensor_d1 >> 16) | (tile_d0 << 16));          // tile_dim0
  g[4] = (int)(tile_d1);                                      // tile_dim1 (dim2=0)
  g[5] = (int)(stride0 & 0xffffffffull);                      // dim0 stride lo
  g[6] = (int)((stride0 >> 32) & 0xffffull);                  // dim0 stride hi
  g[7] = 0;
  return g;
}
__device__ __forceinline__ void tdm_load(v4u g0, v8i g1) {
  v4i z4 = {0, 0, 0, 0};
#if __clang_major__ >= 23
  v8i z8 = {0, 0, 0, 0, 0, 0, 0, 0};
  __builtin_amdgcn_tensor_load_to_lds(g0, g1, z4, z4, z8, 0);
#else
  __builtin_amdgcn_tensor_load_to_lds(g0, g1, z4, z4, 0);
#endif
}
#endif  // HAVE_TDM

// ---------------------------------------------------------------------------
// Kernel 1: LayerNorm + 5 fused projections + gating.
// grid = 2048 blocks x 256 threads; each block owns 128 rows of [N, 128].
// ---------------------------------------------------------------------------
__global__ __launch_bounds__(256)
void ln_proj_kernel(const float* __restrict__ z,
                    const float* __restrict__ ln_w, const float* __restrict__ ln_b,
                    const float* __restrict__ Wga, const float* __restrict__ bga,
                    const float* __restrict__ Wla, const float* __restrict__ bla,
                    const float* __restrict__ Wgb, const float* __restrict__ bgb,
                    const float* __restrict__ Wlb, const float* __restrict__ blb,
                    const float* __restrict__ Wgo, const float* __restrict__ bgo,
                    __bf16* __restrict__ aT, __bf16* __restrict__ bTT,
                    __bf16* __restrict__ gate) {
  extern __shared__ char smem[];
  __bf16* Ws   = (__bf16*)smem;                      // [5][128][128], [c][d]
  float*  bS   = (float*)(smem + 5 * 16384 * 2);     // [5][128]
  float*  lnwS = bS + 5 * 128;                       // [128]
  float*  lnbS = lnwS + 128;                         // [128]
  __bf16* znS  = (__bf16*)(lnbS + 128);              // [16][128]

  const int tid  = threadIdx.x;
  const int wave = tid >> 5;
  const int lane = tid & 31;

  // stage weights transposed (f32 [d][c] -> bf16 [c][d]) and biases into LDS
  const float* wsrc[5] = {Wga, Wla, Wgb, Wlb, Wgo};
  const float* bsrc[5] = {bga, bla, bgb, blb, bgo};
#pragma unroll
  for (int w = 0; w < 5; ++w) {
    const float* src = wsrc[w];
    for (int i = tid; i < 16384; i += 256) {
      const int k = i >> 7, n = i & 127;        // i = k*128+n in source
      Ws[w * 16384 + n * 128 + k] = (__bf16)src[i];
    }
    if (tid < 128) bS[w * 128 + tid] = bsrc[w][tid];
  }
  if (tid < 128) { lnwS[tid] = ln_w[tid]; lnbS[tid] = ln_b[tid]; }
  __syncthreads();

  const int ch = wave * 16 + (lane & 15);   // output channel of this lane
  const int mq = 8 * (lane >> 4);           // C/D row phase

  for (int rt = 0; rt < 8; ++rt) {
    const int rowbase = blockIdx.x * 128 + rt * 16;

    // ---- LayerNorm a 16-row tile into bf16 LDS (16 threads per row) ----
    {
      const int r = tid >> 4, g = tid & 15;
      const float* zp = z + (size_t)(rowbase + r) * DIM + g * 8;
      float v[8];
#pragma unroll
      for (int t = 0; t < 8; ++t) v[t] = zp[t];
      float s = 0.f, ss = 0.f;
#pragma unroll
      for (int t = 0; t < 8; ++t) { s += v[t]; ss += v[t] * v[t]; }
#pragma unroll
      for (int off = 8; off; off >>= 1) {
        s  += __shfl_xor(s,  off, 16);
        ss += __shfl_xor(ss, off, 16);
      }
      const float mu   = s * (1.f / 128.f);
      const float var  = ss * (1.f / 128.f) - mu * mu;
      const float rstd = rsqrtf(var + 1e-5f);
#pragma unroll
      for (int t = 0; t < 8; ++t) {
        const int d = g * 8 + t;
        znS[r * DIM + d] = (__bf16)((v[t] - mu) * rstd * lnwS[d] + lnbS[d]);
      }
    }
    __syncthreads();

    // ---- 5 GEMMs: [16x128] x [128x16] per wave, K in 4 steps of 32 ----
    v8f aga = {}, ala = {}, agb = {}, alb = {}, ago = {};
#pragma unroll
    for (int ks = 0; ks < 4; ++ks) {
      const int k0 = ks * 32;
      v16bf af  = frag_rm(znS, DIM, 0, k0, lane);
      v16bf fga = frag_rm(Ws + 0 * 16384, DIM, wave * 16, k0, lane);
      v16bf fla = frag_rm(Ws + 1 * 16384, DIM, wave * 16, k0, lane);
      v16bf fgb = frag_rm(Ws + 2 * 16384, DIM, wave * 16, k0, lane);
      v16bf flb = frag_rm(Ws + 3 * 16384, DIM, wave * 16, k0, lane);
      v16bf fgo = frag_rm(Ws + 4 * 16384, DIM, wave * 16, k0, lane);
      aga = WMMA_BF16(af, fga, aga);
      ala = WMMA_BF16(af, fla, ala);
      agb = WMMA_BF16(af, fgb, agb);
      alb = WMMA_BF16(af, flb, alb);
      ago = WMMA_BF16(af, fgo, ago);
    }

    // ---- gated epilogue ----
#pragma unroll
    for (int r = 0; r < 8; ++r) {
      const int gr = rowbase + mq + r;          // flat (x,y): x=gr>>9, y=gr&511
      const float av = sigm(aga[r] + bS[0 * 128 + ch]) * (ala[r] + bS[1 * 128 + ch]);
      const float bv = sigm(agb[r] + bS[2 * 128 + ch]) * (alb[r] + bS[3 * 128 + ch]);
      const float gv = sigm(ago[r] + bS[4 * 128 + ch]);
      // a[i=x, m=y, c]  -> aT [c][x][y]
      aT[(size_t)ch * NTOT + gr] = (__bf16)av;
      // b[m=x, j=y, c]  -> bTT[c][y][x]  (N-major for kernel 2's B operand)
      bTT[(size_t)ch * NTOT + (size_t)(gr & 511) * LSEQ + (gr >> 9)] = (__bf16)bv;
      gate[(size_t)gr * DIM + ch] = (__bf16)gv;
    }
    __syncthreads();
  }
}

// ---------------------------------------------------------------------------
// Kernel 2: per-channel 512x512x512 bf16 GEMM (the triangular einsum).
// grid = (32 tiles of 64x128, 128 channels) x 256 threads (8 waves, 2x4).
// Each wave owns a 32x32 output (4 accumulators, A/B fragments shared).
// K-slabs are staged with tensor_load_to_lds (TDM), double-buffered:
// wave 0 issues the next slab's two descriptors, then s_wait_tensorcnt on
// the current slab before the barrier; the next slab's load overlaps all 8 waves' WMMAs.
// ---------------------------------------------------------------------------
__global__ __launch_bounds__(256)
void tri_gemm_kernel(const __bf16* __restrict__ aT,
                     const __bf16* __restrict__ bTT,
                     __bf16* __restrict__ kws) {
  __shared__ __align__(16) __bf16 As[2][64 * 32];    // [m][k]
  __shared__ __align__(16) __bf16 Bs[2][128 * 32];   // [n][k]

  const int c     = blockIdx.y;
  const int mbase = (blockIdx.x >> 2) * 64;       // 8 m-tiles
  const int nbase = (blockIdx.x & 3) * 128;       // 4 n-tiles
  const int tid   = threadIdx.x;
  const int wave  = tid >> 5;
  const int lane  = tid & 31;
  const int wm    = (wave & 1) * 32;              // wave row offset in tile
  const int wn    = (wave >> 1) * 32;             // wave col offset in tile

  const __bf16* Ac = aT  + (size_t)c * NTOT;
  const __bf16* Bc = bTT + (size_t)c * NTOT;

  v8f acc00 = {}, acc01 = {}, acc10 = {}, acc11 = {};

#ifdef HAVE_TDM
  const bool issuer = (wave == 0);
  const v8i g1a = tdm_group1_2d(/*tile*/ 32, 64,  /*tensor*/ 512, 512, /*stride*/ 512);
  const v8i g1b = tdm_group1_2d(/*tile*/ 32, 128, /*tensor*/ 512, 512, /*stride*/ 512);

  if (issuer) {
    tdm_load(tdm_group0((unsigned)(uintptr_t)&As[0][0],
                        (unsigned long long)(uintptr_t)(Ac + (size_t)mbase * LSEQ)), g1a);
    tdm_load(tdm_group0((unsigned)(uintptr_t)&Bs[0][0],
                        (unsigned long long)(uintptr_t)(Bc + (size_t)nbase * LSEQ)), g1b);
  }

  for (int ks = 0; ks < 16; ++ks) {
    const int cur = ks & 1;
    if (issuer) {
      if (ks < 15) {
        const int k1 = (ks + 1) * 32;
        tdm_load(tdm_group0((unsigned)(uintptr_t)&As[cur ^ 1][0],
                            (unsigned long long)(uintptr_t)(Ac + (size_t)mbase * LSEQ + k1)), g1a);
        tdm_load(tdm_group0((unsigned)(uintptr_t)&Bs[cur ^ 1][0],
                            (unsigned long long)(uintptr_t)(Bc + (size_t)nbase * LSEQ + k1)), g1b);
        __builtin_amdgcn_s_wait_tensorcnt(2);   // current slab landed
      } else {
        __builtin_amdgcn_s_wait_tensorcnt(0);
      }
    }
    __syncthreads();

    v16bf a0 = frag_rm(&As[cur][0], 32, wm,      0, lane);
    v16bf a1 = frag_rm(&As[cur][0], 32, wm + 16, 0, lane);
    v16bf b0 = frag_rm(&Bs[cur][0], 32, wn,      0, lane);
    v16bf b1 = frag_rm(&Bs[cur][0], 32, wn + 16, 0, lane);
    acc00 = WMMA_BF16(a0, b0, acc00);
    acc01 = WMMA_BF16(a0, b1, acc01);
    acc10 = WMMA_BF16(a1, b0, acc10);
    acc11 = WMMA_BF16(a1, b1, acc11);
    __syncthreads();
  }
#else
  // Fallback: per-thread staging (no TDM builtin on this toolchain)
  const int ra = tid >> 2, sa = (tid & 3) * 8;
  const int rb1 = ra + 64;
  for (int k0 = 0; k0 < LSEQ; k0 += 32) {
    *(uint4*)(&As[0][0] + ra * 32 + sa) =
        *(const uint4*)(Ac + (size_t)(mbase + ra) * LSEQ + k0 + sa);
    *(uint4*)(&Bs[0][0] + ra * 32 + sa) =
        *(const uint4*)(Bc + (size_t)(nbase + ra) * LSEQ + k0 + sa);
    *(uint4*)(&Bs[0][0] + rb1 * 32 + sa) =
        *(const uint4*)(Bc + (size_t)(nbase + rb1) * LSEQ + k0 + sa);
    __syncthreads();
    v16bf a0 = frag_rm(&As[0][0], 32, wm,      0, lane);
    v16bf a1 = frag_rm(&As[0][0], 32, wm + 16, 0, lane);
    v16bf b0 = frag_rm(&Bs[0][0], 32, wn,      0, lane);
    v16bf b1 = frag_rm(&Bs[0][0], 32, wn + 16, 0, lane);
    acc00 = WMMA_BF16(a0, b0, acc00);
    acc01 = WMMA_BF16(a0, b1, acc01);
    acc10 = WMMA_BF16(a1, b0, acc10);
    acc11 = WMMA_BF16(a1, b1, acc11);
    __syncthreads();
  }
#endif

  // write k[i, j, c] (bf16)
  const int n0 = nbase + wn + (lane & 15);
  const int mq = 8 * (lane >> 4);
#pragma unroll
  for (int r = 0; r < 8; ++r) {
    const int i0 = mbase + wm + mq + r;
    const int i1 = i0 + 16;
    kws[((size_t)i0 * LSEQ + n0) * DIM + c]      = (__bf16)acc00[r];
    kws[((size_t)i0 * LSEQ + n0 + 16) * DIM + c] = (__bf16)acc01[r];
    kws[((size_t)i1 * LSEQ + n0) * DIM + c]      = (__bf16)acc10[r];
    kws[((size_t)i1 * LSEQ + n0 + 16) * DIM + c] = (__bf16)acc11[r];
  }
}

// ---------------------------------------------------------------------------
// Kernel 3: out = gate * (LN_c(k) @ Wlo + blo), f32 output.
// grid = 2048 blocks x 256 threads; 128 rows per block.
// ---------------------------------------------------------------------------
__global__ __launch_bounds__(256)
void out_proj_kernel(const __bf16* __restrict__ kws,
                     const float* __restrict__ ln_w, const float* __restrict__ ln_b,
                     const float* __restrict__ Wlo, const float* __restrict__ blo,
                     const __bf16* __restrict__ gate,
                     float* __restrict__ out) {
  __shared__ __align__(16) __bf16 WloT[128 * 128];   // [d][c]
  __shared__ float bloS[128], lnwS[128], lnbS[128];
  __shared__ __align__(16) __bf16 knS[16 * 128];

  const int tid  = threadIdx.x;
  const int wave = tid >> 5;
  const int lane = tid & 31;

  for (int i = tid; i < 16384; i += 256) {
    const int k = i >> 7, n = i & 127;               // Wlo row-major [c][d]
    WloT[n * 128 + k] = (__bf16)Wlo[i];
  }
  if (tid < 128) { bloS[tid] = blo[tid]; lnwS[tid] = ln_w[tid]; lnbS[tid] = ln_b[tid]; }
  __syncthreads();

  const int dcol = wave * 16 + (lane & 15);
  const int mq   = 8 * (lane >> 4);

  for (int rt = 0; rt < 8; ++rt) {
    const int rowbase = blockIdx.x * 128 + rt * 16;

    // LayerNorm over channels (bf16 in, f32 stats)
    {
      const int r = tid >> 4, g = tid & 15;
      const __bf16* kp = kws + (size_t)(rowbase + r) * DIM + g * 8;
      float v[8];
#pragma unroll
      for (int t = 0; t < 8; ++t) v[t] = (float)kp[t];
      float s = 0.f, ss = 0.f;
#pragma unroll
      for (int t = 0; t < 8; ++t) { s += v[t]; ss += v[t] * v[t]; }
#pragma unroll
      for (int off = 8; off; off >>= 1) {
        s  += __shfl_xor(s,  off, 16);
        ss += __shfl_xor(ss, off, 16);
      }
      const float mu   = s * (1.f / 128.f);
      const float var  = ss * (1.f / 128.f) - mu * mu;
      const float rstd = rsqrtf(var + 1e-5f);
#pragma unroll
      for (int t = 0; t < 8; ++t) {
        const int ci = g * 8 + t;
        knS[r * DIM + ci] = (__bf16)((v[t] - mu) * rstd * lnwS[ci] + lnbS[ci]);
      }
    }
    __syncthreads();

    v8f acc = {};
#pragma unroll
    for (int ks = 0; ks < 4; ++ks) {
      v16bf af = frag_rm(knS, DIM, 0, ks * 32, lane);
      v16bf wf = frag_rm(WloT, DIM, wave * 16, ks * 32, lane);
      acc = WMMA_BF16(af, wf, acc);
    }

#pragma unroll
    for (int r = 0; r < 8; ++r) {
      const int gr = rowbase + mq + r;
      const float res = acc[r] + bloS[dcol];
      const float gv  = (float)gate[(size_t)gr * DIM + dcol];
      out[(size_t)gr * DIM + dcol] = gv * res;
    }
    __syncthreads();
  }
}

// ---------------------------------------------------------------------------
extern "C" void kernel_launch(void* const* d_in, const int* in_sizes, int n_in,
                              void* d_out, int out_size, void* d_ws, size_t ws_size,
                              hipStream_t stream) {
  const float* z       = (const float*)d_in[0];
  const float* ln_in_w = (const float*)d_in[1];
  const float* ln_in_b = (const float*)d_in[2];
  const float* Wga = (const float*)d_in[3];  const float* bga = (const float*)d_in[4];
  const float* Wla = (const float*)d_in[5];  const float* bla = (const float*)d_in[6];
  const float* Wgb = (const float*)d_in[7];  const float* bgb = (const float*)d_in[8];
  const float* Wlb = (const float*)d_in[9];  const float* blb = (const float*)d_in[10];
  const float* ln_o_w = (const float*)d_in[11];
  const float* ln_o_b = (const float*)d_in[12];
  const float* Wgo = (const float*)d_in[13]; const float* bgo = (const float*)d_in[14];
  const float* Wlo = (const float*)d_in[15]; const float* blo = (const float*)d_in[16];
  float* outp = (float*)d_out;

  __bf16* aT   = (__bf16*)((char*)d_ws + OFF_AT);
  __bf16* bTT  = (__bf16*)((char*)d_ws + OFF_BT);
  __bf16* gate = (__bf16*)((char*)d_ws + OFF_GATE);
  __bf16* kws  = (__bf16*)((char*)d_ws + OFF_K);

  // K1: LN + 5 projections + gating (dynamic LDS: 5 bf16 weights + params)
  const size_t smemA = 5 * 16384 * 2 + 5 * 128 * 4 + 2 * 128 * 4 + 16 * 128 * 2;
  ln_proj_kernel<<<NTOT / 128, 256, smemA, stream>>>(
      z, ln_in_w, ln_in_b, Wga, bga, Wla, bla, Wgb, bgb, Wlb, blb, Wgo, bgo,
      aT, bTT, gate);

  // K2: 128 per-channel 512^3 bf16 GEMMs (TDM-staged, double-buffered)
  tri_gemm_kernel<<<dim3(32, 128), 256, 0, stream>>>(aT, bTT, kws);

  // K3: LN over channels + Wlo projection + gate
  out_proj_kernel<<<NTOT / 128, 256, 0, stream>>>(
      kws, ln_o_w, ln_o_b, Wlo, blo, gate, outp);
}
